// HGT_5411658793575
// MI455X (gfx1250) — matrix operand order repaired and checked by reading, not compiled, not run on hardware
//
#include <hip/hip_runtime.h>
#include <hip/hip_bf16.h>
#include <math.h>

// ---------------------------------------------------------------------------
// HGT forward for gfx1250 (MI455X).
//  - Weights pre-packed once per launch to bf16, transposed [N][K]: B-fragment
//    for a K-step is 32 contiguous bytes/lane -> 2x global_load_b128 from L2.
//  - A fragment = two contiguous 8-float runs/lane -> 4x global_load_b128.
//  - One wave computes 1 M-tile x NT N-tiles sharing the A fragment
//    (NT=8: kqv GEMM reads A 3x, out-proj reads A once).  All NT B fragments
//    are preloaded (one clause) before the A convert, then the 8
//    v_wmma_f32_16x16x32_bf16 issue as an uninterrupted burst.
//  - Sparse attention: VALU + float atomics (gather/scatter bound).
// ---------------------------------------------------------------------------

typedef __attribute__((ext_vector_type(16))) __bf16 v16bf;
typedef __attribute__((ext_vector_type(8)))  float  v8f;

#define NHEAD 8
#define HDIM  16
#define HIDC  128

__device__ __forceinline__ float gelu_exact(float x) {
  return 0.5f * x * (1.0f + erff(x * 0.70710678118654752440f));
}

// ---------------------------------------------------------------------------
// Weight pack: Wt[n*K + k] = bf16(W[k*N + n])   (transpose + convert)
// ---------------------------------------------------------------------------
__global__ __launch_bounds__(256)
void hgt_pack_w(const float* __restrict__ W, __bf16* __restrict__ Wt,
                int K, int N, int total) {
  const int t = blockIdx.x * blockDim.x + threadIdx.x;
  if (t >= total) return;
  const int k = t / N, n = t % N;
  Wt[(size_t)n * K + k] = (__bf16)W[t];
}

// ---------------------------------------------------------------------------
// WMMA GEMM: C[M,N] = epi(actA(A[M,K]) @ B[K,N] + bias[N])
//   Bt = bf16 packed [N][K].
//   actA: 0 = identity, 1 = exact GELU on A elements
//   epi : 0 = bias, 1 = bias+ReLU, 2 = skip-blend with sigmoid(*skip_p)
// One wave computes 16 x (16*NT) of C.  M%16==0, (N/16)%NT==0, K%32==0.
// ---------------------------------------------------------------------------
template <int NT>
__global__ __launch_bounds__(128)
void hgt_gemm_wmma(const float* __restrict__ A, const __bf16* __restrict__ Bt,
                   const float* __restrict__ bias, float* __restrict__ C,
                   int M, int N, int K, int actA, int epi,
                   const float* __restrict__ skip_p,
                   const float* __restrict__ Xold) {
  const int gtid    = blockIdx.x * blockDim.x + threadIdx.x;
  const int wave    = gtid >> 5;
  const int lane    = threadIdx.x & 31;
  const int groupsN = (N >> 4) / NT;
  const int tilesM  = M >> 4;
  if (wave >= tilesM * groupsN) return;          // wave-uniform guard (EXEC all-1)
  const int tm   = (wave / groupsN) << 4;
  const int tn0  = (wave % groupsN) * (16 * NT);
  const int m    = lane & 15;                    // row (A) / col (B,C) within tile
  const int half = lane >> 4;

  v8f c[NT];
  #pragma unroll
  for (int t = 0; t < NT; ++t)
    #pragma unroll
    for (int i = 0; i < 8; ++i) c[t][i] = 0.0f;

  const float* Arow = A + (size_t)(tm + m) * K;
  const __bf16* Bl  = Bt + (size_t)(tn0 + m) * K + 16 * half;   // lane's B base

  for (int kk = 0; kk < K; kk += 32) {
    // ---- preload all NT B fragments (independent b128 pairs, one clause);
    //      their latency hides under the A load + GELU/convert below
    v16bf bfr[NT];
    #pragma unroll
    for (int t = 0; t < NT; ++t)
      bfr[t] = *(const v16bf*)(Bl + (size_t)t * 16 * K + kk);
    // ---- A fragment (shared by all NT tiles):
    //      a[0..7]  = A[m][kk + 16*half + 0..7]
    //      a[8..15] = A[m][kk + 16*half + 8 + 0..7]     (4x b128 loads)
    float ar[16];
    {
      const float4* p0 = (const float4*)(Arow + kk + 16 * half);
      const float4* p1 = (const float4*)(Arow + kk + 16 * half + 8);
      float4 r0 = p0[0], r1 = p0[1], r2 = p1[0], r3 = p1[1];
      ar[0] = r0.x; ar[1] = r0.y; ar[2]  = r0.z; ar[3]  = r0.w;
      ar[4] = r1.x; ar[5] = r1.y; ar[6]  = r1.z; ar[7]  = r1.w;
      ar[8] = r2.x; ar[9] = r2.y; ar[10] = r2.z; ar[11] = r2.w;
      ar[12] = r3.x; ar[13] = r3.y; ar[14] = r3.z; ar[15] = r3.w;
    }
    v16bf a;
    #pragma unroll
    for (int i = 0; i < 16; ++i) {
      float x = ar[i];
      if (actA) x = gelu_exact(x);
      a[i] = (__bf16)x;
    }
    // ---- uninterrupted WMMA burst: NT independent accumulators
    #pragma unroll
    for (int t = 0; t < NT; ++t)
      c[t] = __builtin_amdgcn_wmma_f32_16x16x32_bf16(false, a, false, bfr[t],
                                                     (short)0, c[t], false, false);
  }

  float sk = 0.0f;
  if (epi == 2) sk = 1.0f / (1.0f + __expf(-skip_p[0]));
  #pragma unroll
  for (int t = 0; t < NT; ++t) {
    const int col = tn0 + t * 16 + m;
    #pragma unroll
    for (int r = 0; r < 8; ++r) {
      const int row = tm + r + 8 * half;         // C VGPR r -> M=r / r+8 by lane half
      float v = c[t][r] + bias[col];
      if (epi == 1)      v = fmaxf(v, 0.0f);
      else if (epi == 2) v = sk * v + (1.0f - sk) * Xold[(size_t)row * HIDC + col];
      C[(size_t)row * N + col] = v;
    }
  }
}

// ---------------------------------------------------------------------------
// Softmax scratch init: amax = ord(-inf), denom = 0, agg = 0
// ---------------------------------------------------------------------------
__global__ __launch_bounds__(256)
void hgt_init_scratch(int* __restrict__ amax, float* __restrict__ denom,
                      float* __restrict__ agg, int n8, int n128) {
  const int t = blockIdx.x * blockDim.x + threadIdx.x;
  if (t < n8)  { amax[t] = (int)0x807fffff; denom[t] = 0.0f; }
  if (t < n128)  agg[t]  = 0.0f;
}

// ordered-int encoding of float for atomicMax
__device__ __forceinline__ int f2ord(float f) {
  int i = __float_as_int(f);
  return (i >= 0) ? i : (i ^ 0x7fffffff);
}
__device__ __forceinline__ float ord2f(int i) {
  return __int_as_float((i >= 0) ? i : (i ^ 0x7fffffff));
}

// ---------------------------------------------------------------------------
// Pass A: per (edge, head):  a = (q[dst] . (k[src] @ Wk[h])) * p[h] / 4
// KQV row layout per node: [k(128) | q(128) | v(128)]
// ---------------------------------------------------------------------------
__global__ __launch_bounds__(256)
void hgt_attn_logits(const int* __restrict__ src, const int* __restrict__ dst,
                     int E, const float* __restrict__ KQVs,
                     const float* __restrict__ KQVd,
                     const float* __restrict__ Wk, const float* __restrict__ prio,
                     float* __restrict__ alpha, int* __restrict__ amax,
                     int dstBase) {
  const int t = blockIdx.x * blockDim.x + threadIdx.x;
  if (t >= E * NHEAD) return;
  const int e = t >> 3, h = t & 7;
  const int s = src[e], d = dst[e];
  const float* krow = KQVs + (size_t)s * 384 + h * HDIM;
  const float* qrow = KQVd + (size_t)d * 384 + 128 + h * HDIM;
  const float* W    = Wk + h * 256;
  float kr[16], qr[16];
  #pragma unroll
  for (int i = 0; i < 16; ++i) { kr[i] = krow[i]; qr[i] = qrow[i]; }
  float acc = 0.0f;
  #pragma unroll
  for (int eo = 0; eo < 16; ++eo) {
    float kt = 0.0f;
    #pragma unroll
    for (int dd = 0; dd < 16; ++dd) kt = fmaf(kr[dd], W[dd * 16 + eo], kt);
    acc = fmaf(kt, qr[eo], acc);
  }
  const float a = acc * prio[h] * 0.25f;         // 1/sqrt(16)
  alpha[(size_t)e * NHEAD + h] = a;
  atomicMax(&amax[(dstBase + d) * NHEAD + h], f2ord(a));
}

// ---------------------------------------------------------------------------
// Pass B: e = exp(a - amax[dst]); alpha <- e; denom += e
// ---------------------------------------------------------------------------
__global__ __launch_bounds__(256)
void hgt_attn_exp(const int* __restrict__ dst, int E,
                  float* __restrict__ alpha, const int* __restrict__ amax,
                  float* __restrict__ denom, int dstBase) {
  const int t = blockIdx.x * blockDim.x + threadIdx.x;
  if (t >= E * NHEAD) return;
  const int e = t >> 3, h = t & 7;
  const int d = dst[e];
  float mx = ord2f(amax[(dstBase + d) * NHEAD + h]);
  if (!(mx > -1e30f)) mx = 0.0f;                 // isfinite guard as in reference
  const float ex = __expf(alpha[(size_t)e * NHEAD + h] - mx);
  alpha[(size_t)e * NHEAD + h] = ex;
  atomicAdd(&denom[(dstBase + d) * NHEAD + h], ex);
}

// ---------------------------------------------------------------------------
// Pass C: w = e/denom; agg[dst] += (v[src] @ Wv[h]) * w   (16 float atomics)
// ---------------------------------------------------------------------------
__global__ __launch_bounds__(256)
void hgt_attn_agg(const int* __restrict__ src, const int* __restrict__ dst,
                  int E, const float* __restrict__ KQVs,
                  const float* __restrict__ Wv, const float* __restrict__ alpha,
                  const float* __restrict__ denom, float* __restrict__ agg,
                  int dstBase) {
  const int t = blockIdx.x * blockDim.x + threadIdx.x;
  if (t >= E * NHEAD) return;
  const int e = t >> 3, h = t & 7;
  const int s = src[e], d = dst[e];
  const float w = alpha[(size_t)e * NHEAD + h] /
                  denom[(dstBase + d) * NHEAD + h];
  const float* vrow = KQVs + (size_t)s * 384 + 256 + h * HDIM;
  const float* W    = Wv + h * 256;
  float vr[16];
  #pragma unroll
  for (int i = 0; i < 16; ++i) vr[i] = vrow[i];
  float* out = agg + (size_t)(dstBase + d) * HIDC + h * HDIM;
  #pragma unroll
  for (int eo = 0; eo < 16; ++eo) {
    float vt = 0.0f;
    #pragma unroll
    for (int dd = 0; dd < 16; ++dd) vt = fmaf(vr[dd], W[dd * 16 + eo], vt);
    atomicAdd(&out[eo], vt * w);
  }
}

// ---------------------------------------------------------------------------
// Host orchestration
// ---------------------------------------------------------------------------
extern "C" void kernel_launch(void* const* d_in, const int* in_sizes, int n_in,
                              void* d_out, int out_size, void* d_ws, size_t ws_size,
                              hipStream_t stream) {
  const int NA = 40000, NP = 80000, NT_ = NA + NP;
  const int EW = 200000, ER = 200000, EC = 150000;

  const float* xa_in = (const float*)d_in[0];
  const float* xp_in = (const float*)d_in[1];
  const int* ei_w = (const int*)d_in[2];
  const int* ei_r = (const int*)d_in[3];
  const int* ei_c = (const int*)d_in[4];
  auto P = [&](int i) { return (const float*)d_in[i]; };
  // param leaf order (jax sorted-key flatten):
  //   layers[l] (base=5+19l): kqv{a:W,b p:W,b} out{a:W,b p:W,b}
  //     rel{cites:Wk,Wv,p rev_writes:Wk,Wv,p writes:Wk,Wv,p} skip{a,p}
  //   lin_in: 43..46 (aW,ab,pW,pb)   lin_out: 47(W),48(b)

  float* ws    = (float*)d_ws;
  float* X0    = ws;
  float* X1    = X0 + (size_t)NT_ * 128;
  float* KQV   = X1 + (size_t)NT_ * 128;
  float* ALPHA = KQV + (size_t)NT_ * 384;
  int*   AMAX  = (int*)(ALPHA + (size_t)(EW + ER + EC) * NHEAD);
  float* DENOM = (float*)(AMAX + (size_t)NT_ * NHEAD);
  float* AGG   = DENOM + (size_t)NT_ * NHEAD;
  __bf16* WP   = (__bf16*)(AGG + (size_t)NT_ * 128);   // packed bf16 weights

  float* KQV_a = KQV;
  float* KQV_p = KQV + (size_t)NA * 384;
  float* AL_w  = ALPHA;
  float* AL_r  = ALPHA + (size_t)EW * NHEAD;
  float* AL_c  = ALPHA + (size_t)(EW + ER) * NHEAD;

  // ---- pack all GEMM weights to bf16 [N][K] once per launch ----
  __bf16* wp_cursor = WP;
  auto packW = [&](const float* W, int K, int N) -> __bf16* {
    __bf16* dst = wp_cursor;
    wp_cursor += (size_t)K * N;
    const int total = K * N;
    hipLaunchKernelGGL(hgt_pack_w, dim3((total + 255) / 256), dim3(256), 0,
                       stream, W, dst, K, N, total);
    return dst;
  };
  __bf16* wp_liA = packW(P(43), 128, 128);
  __bf16* wp_liP = packW(P(45), 64, 128);
  __bf16* wp_kqvA[2], *wp_kqvP[2], *wp_outA[2], *wp_outP[2];
  for (int l = 0; l < 2; ++l) {
    const int b = 5 + 19 * l;
    wp_kqvA[l] = packW(P(b + 0), 128, 384);
    wp_kqvP[l] = packW(P(b + 2), 128, 384);
    wp_outA[l] = packW(P(b + 4), 128, 128);
    wp_outP[l] = packW(P(b + 6), 128, 128);
  }
  __bf16* wp_lo = packW(P(47), 128, 16);

  auto gemm = [&](const float* A, const __bf16* Bt, const float* bias, float* C,
                  int M, int N, int K, int actA, int epi,
                  const float* skip, const float* Xold) {
    if ((N / 16) % 8 == 0) {
      const int waves  = (M / 16) * (N / 128);   // NT = 8
      const int blocks = (waves + 3) / 4;        // 128 threads = 4 waves
      hipLaunchKernelGGL(HIP_KERNEL_NAME(hgt_gemm_wmma<8>), dim3(blocks),
                         dim3(128), 0, stream,
                         A, Bt, bias, C, M, N, K, actA, epi, skip, Xold);
    } else {
      const int waves  = (M / 16) * (N / 16);    // NT = 1
      const int blocks = (waves + 3) / 4;
      hipLaunchKernelGGL(HIP_KERNEL_NAME(hgt_gemm_wmma<1>), dim3(blocks),
                         dim3(128), 0, stream,
                         A, Bt, bias, C, M, N, K, actA, epi, skip, Xold);
    }
  };

  float* xcur = X0;
  float* xnxt = X1;

  // input projections + ReLU
  gemm(xa_in, wp_liA, P(44), xcur,                    NA, 128, 128, 0, 1, nullptr, nullptr);
  gemm(xp_in, wp_liP, P(46), xcur + (size_t)NA * 128, NP, 128, 64,  0, 1, nullptr, nullptr);

  for (int l = 0; l < 2; ++l) {
    const int b = 5 + 19 * l;
    // K/Q/V projections
    gemm(xcur,                    wp_kqvA[l], P(b + 1), KQV_a, NA, 384, 128, 0, 0, nullptr, nullptr);
    gemm(xcur + (size_t)NA * 128, wp_kqvP[l], P(b + 3), KQV_p, NP, 384, 128, 0, 0, nullptr, nullptr);

    // softmax scratch init
    const int n128 = NT_ * 128, n8 = NT_ * NHEAD;
    hipLaunchKernelGGL(hgt_init_scratch, dim3((n128 + 255) / 256), dim3(256), 0,
                       stream, AMAX, DENOM, AGG, n8, n128);

    // pass A: logits + segment max     (writes: a->p, rev: p->a, cites: p->p)
    hipLaunchKernelGGL(hgt_attn_logits, dim3((EW * NHEAD + 255) / 256), dim3(256), 0, stream,
                       ei_w, ei_w + EW, EW, KQV_a, KQV_p, P(b + 14), P(b + 16), AL_w, AMAX, NA);
    hipLaunchKernelGGL(hgt_attn_logits, dim3((ER * NHEAD + 255) / 256), dim3(256), 0, stream,
                       ei_r, ei_r + ER, ER, KQV_p, KQV_a, P(b + 11), P(b + 13), AL_r, AMAX, 0);
    hipLaunchKernelGGL(hgt_attn_logits, dim3((EC * NHEAD + 255) / 256), dim3(256), 0, stream,
                       ei_c, ei_c + EC, EC, KQV_p, KQV_p, P(b + 8),  P(b + 10), AL_c, AMAX, NA);

    // pass B: exp + segment sum
    hipLaunchKernelGGL(hgt_attn_exp, dim3((EW * NHEAD + 255) / 256), dim3(256), 0, stream,
                       ei_w + EW, EW, AL_w, AMAX, DENOM, NA);
    hipLaunchKernelGGL(hgt_attn_exp, dim3((ER * NHEAD + 255) / 256), dim3(256), 0, stream,
                       ei_r + ER, ER, AL_r, AMAX, DENOM, 0);
    hipLaunchKernelGGL(hgt_attn_exp, dim3((EC * NHEAD + 255) / 256), dim3(256), 0, stream,
                       ei_c + EC, EC, AL_c, AMAX, DENOM, NA);

    // pass C: weighted message scatter
    hipLaunchKernelGGL(hgt_attn_agg, dim3((EW * NHEAD + 255) / 256), dim3(256), 0, stream,
                       ei_w, ei_w + EW, EW, KQV_a, P(b + 15), AL_w, DENOM, AGG, NA);
    hipLaunchKernelGGL(hgt_attn_agg, dim3((ER * NHEAD + 255) / 256), dim3(256), 0, stream,
                       ei_r, ei_r + ER, ER, KQV_p, P(b + 12), AL_r, DENOM, AGG, 0);
    hipLaunchKernelGGL(hgt_attn_agg, dim3((EC * NHEAD + 255) / 256), dim3(256), 0, stream,
                       ei_c, ei_c + EC, EC, KQV_p, P(b + 9),  AL_c, DENOM, AGG, NA);

    // output projection: gelu(agg) @ W + b, skip-blend with old x
    gemm(AGG,                    wp_outA[l], P(b + 5), xnxt,                    NA, 128, 128, 1, 2,
         P(b + 17), xcur);
    gemm(AGG + (size_t)NA * 128, wp_outP[l], P(b + 7), xnxt + (size_t)NA * 128, NP, 128, 128, 1, 2,
         P(b + 18), xcur + (size_t)NA * 128);

    float* tmp = xcur; xcur = xnxt; xnxt = tmp;
  }

  // final head: author features -> [40000, 16]
  gemm(xcur, wp_lo, P(48), (float*)d_out, NA, 16, 128, 0, 0, nullptr, nullptr);
}
